// DigitCapsules_20194936226663
// MI455X (gfx1250) — matrix operand order repaired and checked by reading, not compiled, not run on hardware
//
#include <hip/hip_runtime.h>

typedef __attribute__((ext_vector_type(16))) _Float16 v16h;
typedef __attribute__((ext_vector_type(8)))  float    v8f;

#define B_   256
#define R_   1152
#define N_   10
#define C_   32
#define O_   16
#define NO_  160
#define EPSQ 1e-8f

// ---------------------------------------------------------------------------
// Pack x (f32, [B,R,C]) into per-wave WMMA A fragments (f16).
// Fragment layout per (btile, r): 32 lanes x 16 halves, contiguous per lane,
// contiguous K order (lane half selects K base 0/16, element e adds e),
// matching the B packing.  One thread per (btile, r, lane).
// ---------------------------------------------------------------------------
__global__ void pack_x_kernel(const float* __restrict__ x,
                              _Float16* __restrict__ xp) {
  int t = blockIdx.x * blockDim.x + threadIdx.x;       // (bt*R + r)*32 + lane
  if (t >= 16 * R_ * 32) return;
  int lane = t & 31;
  int rem  = t >> 5;                                   // bt*R + r
  int r    = rem % R_;
  int bt   = rem / R_;
  int m    = lane & 15;
  int hi   = (lane >> 4) << 4;                         // 0 or 16
  const float* src = x + ((size_t)(bt * 16 + m) * R_ + r) * C_;
  v16h out;
#pragma unroll
  for (int e = 0; e < 16; ++e) {
    int cc = hi + e;
    out[e] = (_Float16)src[cc];
  }
  *(v16h*)(xp + (size_t)t * 16) = out;                 // coalesced 32B/lane
}

// ---------------------------------------------------------------------------
// Pack W (f32, [R,N,C,O]) into WMMA B fragments (f16): 32x16 layout,
// column = lane&15, K base 0/16 by lane half.  One thread per (r, n, lane).
// ---------------------------------------------------------------------------
__global__ void pack_W_kernel(const float* __restrict__ W,
                              _Float16* __restrict__ Wp) {
  int t = blockIdx.x * blockDim.x + threadIdx.x;       // (r*N + n)*32 + lane
  if (t >= R_ * N_ * 32) return;
  int lane = t & 31;
  int rn   = t >> 5;
  int o    = lane & 15;
  int cb   = (lane >> 4) << 4;                         // 0 or 16
  const float* src = W + ((size_t)rn * C_ + cb) * O_ + o;
  v16h out;
#pragma unroll
  for (int e = 0; e < 16; ++e) out[e] = (_Float16)src[(size_t)e * O_];
  *(v16h*)(Wp + (size_t)t * 16) = out;
}

// ---------------------------------------------------------------------------
// u_hat[b,r,n,o] = sum_c x[b,r,c]*W[r,n,c,o].
// One wave per (r, n); B fragment loaded once, 16 WMMAs over batch tiles.
// Inner loop: 2x b128 A load + 1 WMMA + 8 coalesced b32 stores.
// ---------------------------------------------------------------------------
__global__ __launch_bounds__(256) void uhat_wmma_kernel(
    const _Float16* __restrict__ xp, const _Float16* __restrict__ Wp,
    float* __restrict__ uhat) {
  const int lane = threadIdx.x & 31;
  const int wave = threadIdx.x >> 5;
  const int gw   = blockIdx.x * 8 + wave;              // 0 .. R_*N_-1
  const int r     = gw / N_;
  const int ntile = gw - r * N_;

  const v16h bm = *(const v16h*)(Wp + ((size_t)gw * 32 + lane) * 16);

  const int o  = lane & 15;
  const int mo = (lane >> 4) << 3;                     // 0 or 8
  float* ub = uhat + (size_t)r * NO_ + ntile * O_ + o; // b=0 base

#pragma unroll
  for (int bt = 0; bt < 16; ++bt) {
    v16h a = *(const v16h*)(xp + (((size_t)bt * R_ + r) * 32 + lane) * 16);
    v8f c = {};
    v8f d = __builtin_amdgcn_wmma_f32_16x16x32_f16(
        false, a, false, bm, (short)0, c, false, false);
#pragma unroll
    for (int g = 0; g < 8; ++g) {
      int bb = bt * 16 + g + mo;
      ub[(size_t)bb * (R_ * NO_)] = d[g];
    }
  }
}

// ---------------------------------------------------------------------------
// softmax over n (N=10) of routing logits -> coupling coefficients c
// ---------------------------------------------------------------------------
__global__ void softmax_kernel(const float* __restrict__ logits,
                               float* __restrict__ c) {
  int idx = blockIdx.x * blockDim.x + threadIdx.x;     // b*R + r
  if (idx >= B_ * R_) return;
  const float* l = logits + (size_t)idx * N_;
  float buf[N_];
  float m = -3.4e38f;
#pragma unroll
  for (int n = 0; n < N_; ++n) { buf[n] = l[n]; m = fmaxf(m, buf[n]); }
  float sum = 0.f;
#pragma unroll
  for (int n = 0; n < N_; ++n) { buf[n] = __expf(buf[n] - m); sum += buf[n]; }
  float inv = 1.f / sum;
  float* co = c + (size_t)idx * N_;
#pragma unroll
  for (int n = 0; n < N_; ++n) co[n] = buf[n] * inv;
}

// ---------------------------------------------------------------------------
// s[b,n,o] = sum_r c[b,r,n] * u_hat[b,r,n,o]   (c==null -> uniform 1/N)
// ---------------------------------------------------------------------------
__global__ __launch_bounds__(160) void reduce_s_kernel(
    const float* __restrict__ uhat, const float* __restrict__ c,
    float* __restrict__ s, int use_c) {
  const int b  = blockIdx.x;
  const int r0 = blockIdx.y * 144;
  const int no = threadIdx.x;
  const int n  = no >> 4;
  float acc = 0.f;
  for (int rr = 0; rr < 144; ++rr) {
    int r = r0 + rr;
    __builtin_prefetch(&uhat[((size_t)b * R_ + r + 8) * NO_ + no], 0, 1);
    float cv = use_c ? c[((size_t)b * R_ + r) * N_ + n] : 0.1f;
    acc += cv * uhat[((size_t)b * R_ + r) * NO_ + no];
  }
  atomicAdd(&s[(size_t)b * NO_ + no], acc);
}

// ---------------------------------------------------------------------------
// v = squash(s): one thread per (b, n)
// ---------------------------------------------------------------------------
__global__ void squash_kernel(const float* __restrict__ s,
                              float* __restrict__ v) {
  int idx = blockIdx.x * blockDim.x + threadIdx.x;     // b*N + n
  if (idx >= B_ * N_) return;
  const float* sp = s + (size_t)idx * O_;
  float t[O_];
  float sq = 0.f;
#pragma unroll
  for (int o = 0; o < O_; ++o) { t[o] = sp[o]; sq += t[o] * t[o]; }
  float scale = (sq / (1.f + sq)) * rsqrtf(sq + EPSQ);
  float* vp = v + (size_t)idx * O_;
#pragma unroll
  for (int o = 0; o < O_; ++o) vp[o] = scale * t[o];
}

// ---------------------------------------------------------------------------
// logits[b,r,n] (+)= dot(u_hat[b,r,n,:], v[b,n,:])
// ---------------------------------------------------------------------------
__global__ void agreement_kernel(const float* __restrict__ uhat,
                                 const float* __restrict__ v,
                                 float* __restrict__ logits, int accumulate) {
  int idx = blockIdx.x * blockDim.x + threadIdx.x;     // ((b*R)+r)*N + n
  if (idx >= B_ * R_ * N_) return;
  int n = idx % N_;
  int b = idx / (R_ * N_);
  const float4* u4 = (const float4*)(uhat + (size_t)idx * O_);
  const float4* v4 = (const float4*)(v + ((size_t)b * N_ + n) * O_);
  float dot = 0.f;
#pragma unroll
  for (int k = 0; k < 4; ++k) {
    float4 a = u4[k], w = v4[k];
    dot += a.x * w.x + a.y * w.y + a.z * w.z + a.w * w.w;
  }
  logits[idx] = (accumulate ? logits[idx] : 0.f) + dot;
}

// ---------------------------------------------------------------------------
extern "C" void kernel_launch(void* const* d_in, const int* in_sizes, int n_in,
                              void* d_out, int out_size, void* d_ws,
                              size_t ws_size, hipStream_t stream) {
  (void)in_sizes; (void)n_in; (void)out_size; (void)ws_size;
  const float* x = (const float*)d_in[0];   // (256,1152,32)
  const float* W = (const float*)d_in[1];   // (1152,10,32,16)
  float* out = (float*)d_out;               // (256,10,16)

  char* ws = (char*)d_ws;
  const size_t UHAT_B   = (size_t)B_ * R_ * NO_ * 4;   // 188,743,680
  const size_t LOGITS_B = (size_t)B_ * R_ * N_ * 4;    //  11,796,480
  const size_t S_B      = (size_t)B_ * NO_ * 4;        //     163,840
  const size_t XP_B     = (size_t)B_ * R_ * C_ * 2;    //  18,874,368
  const size_t WP_B     = (size_t)R_ * N_ * C_ * O_ * 2; // 11,796,480

  size_t off = 0;
  float*    uhat   = (float*)(ws + off);    off += UHAT_B;
  float*    logits = (float*)(ws + off);    off += LOGITS_B;
  float*    cbuf   = (float*)(ws + off);    off += LOGITS_B;
  float*    s      = (float*)(ws + off);    off += S_B;
  float*    v      = (float*)(ws + off);    off += S_B;
  _Float16* xp     = (_Float16*)(ws + off); off += XP_B;
  _Float16* Wp     = (_Float16*)(ws + off); off += WP_B;

  const int nAgree = (B_ * R_ * N_ + 255) / 256;
  const int nSmax  = (B_ * R_ + 255) / 256;

  // one-shot operand packing into WMMA fragment layout (f16)
  pack_x_kernel<<<(16 * R_ * 32 + 255) / 256, 256, 0, stream>>>(x, xp);
  pack_W_kernel<<<(R_ * N_ * 32 + 255) / 256, 256, 0, stream>>>(W, Wp);

  // u_hat GEMM: one wave per (r, n), 16 WMMAs over batch tiles
  uhat_wmma_kernel<<<(R_ * N_) / 8, 256, 0, stream>>>(xp, Wp, uhat);

  // ---- iteration 1 (logits=0 -> uniform coupling 1/N) ----
  hipMemsetAsync(s, 0, S_B, stream);
  reduce_s_kernel<<<dim3(B_, 8), 160, 0, stream>>>(uhat, nullptr, s, 0);
  squash_kernel<<<10, 256, 0, stream>>>(s, v);
  agreement_kernel<<<nAgree, 256, 0, stream>>>(uhat, v, logits, 0);

  // ---- iteration 2 ----
  softmax_kernel<<<nSmax, 256, 0, stream>>>(logits, cbuf);
  hipMemsetAsync(s, 0, S_B, stream);
  reduce_s_kernel<<<dim3(B_, 8), 160, 0, stream>>>(uhat, cbuf, s, 1);
  squash_kernel<<<10, 256, 0, stream>>>(s, v);
  agreement_kernel<<<nAgree, 256, 0, stream>>>(uhat, v, logits, 1);

  // ---- iteration 3 (final v -> d_out) ----
  softmax_kernel<<<nSmax, 256, 0, stream>>>(logits, cbuf);
  hipMemsetAsync(s, 0, S_B, stream);
  reduce_s_kernel<<<dim3(B_, 8), 160, 0, stream>>>(uhat, cbuf, s, 1);
  squash_kernel<<<10, 256, 0, stream>>>(s, out);
}
